// LSNNPolicy_54589034332671
// MI455X (gfx1250) — matrix operand order repaired and checked by reading, not compiled, not run on hardware
//
#include <hip/hip_runtime.h>
#include <hip/hip_bf16.h>

// CDNA5 (gfx1250) LSNN forward: whole 40-step scan fused in one kernel.
// Recurrent matmul z @ w_rec^T runs on v_wmma_f32_16x16x32_f16 with the
// synaptic-current state as the accumulator; all neuron state lives in
// WMMA C-layout register fragments; spikes ping-pong through LDS in f16.

typedef _Float16 v16h __attribute__((ext_vector_type(16)));
typedef _Float16 v8h  __attribute__((ext_vector_type(8)));
typedef float    v8f  __attribute__((ext_vector_type(8)));
typedef float    v4f  __attribute__((ext_vector_type(4)));

#define SEQ_LEN 40
#define HID     128
#define NFEAT   8     // 4 pos + 4 neg encoded channels
#define ROWS    16    // batch rows per workgroup (one WMMA M tile)
#define NT      2     // 16-wide N tiles per wave  (wave owns 32 neurons)
#define KC      4     // 32-wide K chunks covering HID=128

// dt=1e-3 Norse defaults
#define DEC_V   0.9f                 // 1 - dt/tau_mem
#define IN_V    0.1f                 // dt/tau_mem
#define DEC_I   0.8f                 // 1 - dt/tau_syn
#define C_BDEC  1.4285714285714286e-6f  // dt * tau_adapt_inv
#define C_BJMP  2.5714285714285714e-3f  // tau_adapt_inv * beta
#define V_TH    1.0f

__global__ __launch_bounds__(128)
void lsnn_fused_kernel(const float* __restrict__ x,      // [B,4]
                       const float* __restrict__ w_in,   // [128,8]
                       const float* __restrict__ w_rec,  // [128,128]
                       const float* __restrict__ w_out,  // [2,128]
                       float* __restrict__ out)          // [B,2] softmax
{
    __shared__ __align__(16) _Float16 zbuf[2][ROWS * HID]; // spike ping-pong, f16
    __shared__ __align__(16) float    xt[ROWS * NFEAT];    // encoded input spikes
    __shared__ __align__(16) float    wout[2 * HID];

    const int tid  = threadIdx.x;
    const int wave = tid >> 5;
    const int lane = tid & 31;
    const int hi   = lane >> 4;   // half-wave select (WMMA layouts)
    const int col  = lane & 15;   // N-in-tile (B/C/D) and M-row (A)
    const int blk  = blockIdx.x;

    // w_out -> LDS (256 floats, 128 threads)
    wout[tid]       = w_out[tid];
    wout[tid + 128] = w_out[tid + 128];

    // ---- Preload constant w_rec^T B-fragments into VGPRs (f32 -> f16).
    // B operand, contiguous k order chosen here (same as A): lane l holds column N = l&15, K = 16*(l>>4)+h.
    // B[k][n] = w_rec[n][k]  => lane reads 16 contiguous f32 of row n.
    v16h wrec[NT][KC];
    #pragma unroll
    for (int t = 0; t < NT; ++t) {
        const int n = wave * 32 + t * 16 + col;
        #pragma unroll
        for (int c = 0; c < KC; ++c) {
            const float* src = w_rec + n * HID + c * 32 + 16 * hi;
            v16h f;
            #pragma unroll
            for (int j = 0; j < 16; ++j) f[j] = (_Float16)src[j];
            wrec[t][c] = f;
        }
    }

    // w_in row for this lane's two neurons (input projection in VALU, K=8 only)
    float win[NT][NFEAT];
    #pragma unroll
    for (int t = 0; t < NT; ++t) {
        const int n = wave * 32 + t * 16 + col;
        #pragma unroll
        for (int f = 0; f < NFEAT; ++f) win[t][f] = w_in[n * NFEAT + f];
    }

    // ---- Inline constant-current LIF encoder: one state per thread.
    const int  erow  = tid >> 3;
    const int  efeat = tid & 7;
    const float xval = x[(blk * ROWS + erow) * 4 + (efeat & 3)];
    const float cur  = fmaxf(0.0f, (efeat < 4 ? 50.0f : -50.0f) * xval);
    float ve = 0.0f;

    // ---- LSNN state in WMMA C-layout fragments (v8f: row = r + 8*hi, n = col)
    v8f vS[NT], iS[NT], bS[NT];
    #pragma unroll
    for (int t = 0; t < NT; ++t)
        #pragma unroll
        for (int r = 0; r < 8; ++r) { vS[t][r] = 0.0f; iS[t][r] = 0.0f; bS[t][r] = V_TH; }

    // z(t-1) A-fragments (16-bit A 16x32 layout), start at zero spikes
    v16h aZ[KC];
    #pragma unroll
    for (int c = 0; c < KC; ++c)
        #pragma unroll
        for (int j = 0; j < 16; ++j) aZ[c][j] = (_Float16)0.0f;

    // readout state (wave 0: lane = row*2 + o, exactly 32 lanes)
    float vo = 0.0f, io = 0.0f, maxv = -1e30f;

    for (int step = 0; step < SEQ_LEN; ++step) {
        const int buf = step & 1;

        // ================= phase A: local dynamics =================
        // encoder step -> xt spikes
        ve = DEC_V * ve + IN_V * cur;
        float sp = (ve - V_TH > 0.0f) ? 1.0f : 0.0f;
        ve = (1.0f - sp) * ve;
        xt[tid] = sp;

        // membrane / threshold update; emit z(t) to LDS as f16
        #pragma unroll
        for (int t = 0; t < NT; ++t) {
            const int n = wave * 32 + t * 16 + col;
            #pragma unroll
            for (int r = 0; r < 8; ++r) {
                const float vdec = DEC_V * vS[t][r] + IN_V * iS[t][r];
                const float bdec = bS[t][r] + C_BDEC * (1.0f - bS[t][r]);
                const float z    = (vdec - bdec > 0.0f) ? 1.0f : 0.0f;
                vS[t][r] = (1.0f - z) * vdec;
                bS[t][r] = bdec + z * C_BJMP;
                const int row = r + 8 * hi;
                zbuf[buf][row * HID + n] = (_Float16)z;
            }
        }
        __syncthreads();

        // ================= phase B: synaptic + readout =================
        // i decay
        #pragma unroll
        for (int t = 0; t < NT; ++t)
            #pragma unroll
            for (int r = 0; r < 8; ++r) iS[t][r] *= DEC_I;

        // input projection: i += x_t @ w_in^T   (K=8, VALU)
        #pragma unroll
        for (int r = 0; r < 8; ++r) {
            const int row = r + 8 * hi;
            const float* xr = &xt[row * NFEAT];
            const v4f x0 = *(const v4f*)xr;
            const v4f x1 = *(const v4f*)(xr + 4);
            #pragma unroll
            for (int t = 0; t < NT; ++t) {
                float acc = 0.0f;
                #pragma unroll
                for (int f = 0; f < 4; ++f) {
                    acc = fmaf(x0[f], win[t][f],     acc);
                    acc = fmaf(x1[f], win[t][4 + f], acc);
                }
                iS[t][r] += acc;
            }
        }

        // recurrent: i += z(t-1) @ w_rec^T   (WMMA f16 -> f32, acc chained in C)
        #pragma unroll
        for (int t = 0; t < NT; ++t) {
            v8f c = iS[t];
            #pragma unroll
            for (int k = 0; k < KC; ++k)
                c = __builtin_amdgcn_wmma_f32_16x16x32_f16(
                        false, aZ[k], false, wrec[t][k], (short)0, c, false, false);
            iS[t] = c;
        }

        // gather z(t) A-fragments for next step.
        // A 16x32 f16 layout: lane holds row=col; halves 0..15 -> K = 32c + 16*hi + j
        // (contiguous, matching B; two aligned 16B LDS loads).
        #pragma unroll
        for (int k = 0; k < KC; ++k) {
            const _Float16* zp = &zbuf[buf][col * HID + k * 32 + 16 * hi];
            v16h a;
            *(v8h*)&a       = *(const v8h*)zp;
            *((v8h*)&a + 1) = *(const v8h*)(zp + 8);
            aZ[k] = a;
        }

        // LI readout (wave 0): vo from OLD io, io from z(t) @ w_out^T
        if (wave == 0) {
            const int row = lane >> 1;
            const int o   = lane & 1;
            float acc = 0.0f;
            const _Float16* zr = &zbuf[buf][row * HID];
            const float*    wr = &wout[o * HID];
            #pragma unroll
            for (int j = 0; j < 16; ++j) {
                const v8h z8 = *(const v8h*)(zr + j * 8);
                #pragma unroll
                for (int e = 0; e < 8; ++e)
                    acc = fmaf((float)z8[e], wr[j * 8 + e], acc);
            }
            vo = DEC_V * vo + IN_V * io;
            io = DEC_I * io + acc;
            maxv = fmaxf(maxv, vo);
        }
        __syncthreads();
    }

    // softmax over O=2 via lane-pair exchange; coalesced store (index == lane)
    if (wave == 0) {
        const float other = __shfl_xor(maxv, 1, 32);
        const float mx = fmaxf(maxv, other);
        const float e  = __expf(maxv - mx);
        const float eo = __expf(other - mx);
        out[blk * 32 + lane] = e / (e + eo);
    }
}

extern "C" void kernel_launch(void* const* d_in, const int* in_sizes, int n_in,
                              void* d_out, int out_size, void* d_ws, size_t ws_size,
                              hipStream_t stream) {
    (void)n_in; (void)out_size; (void)d_ws; (void)ws_size;
    const float* x     = (const float*)d_in[0];  // [B,4]
    const float* w_in  = (const float*)d_in[1];  // [128,8]
    const float* w_rec = (const float*)d_in[2];  // [128,128]
    const float* w_out = (const float*)d_in[3];  // [2,128]
    float* out = (float*)d_out;

    const int B = in_sizes[0] / 4;
    const int blocks = B / ROWS;   // 4096 for B=65536
    lsnn_fused_kernel<<<blocks, 128, 0, stream>>>(x, w_in, w_rec, w_out, out);
}
